// MoERouter_7894149890290
// MI455X (gfx1250) — matrix operand order repaired and checked by reading, not compiled, not run on hardware
//
#include <hip/hip_runtime.h>

// MoE router fused kernel for MI455X (gfx1250, wave32).
// Single pass over x (268 MB -> HBM-bound ~11.5us @ 23.3 TB/s):
//   - gate_w * router_scale staged once per WG into LDS as f16 (257 KB)
//   - per-wave 16x64 logits via V_WMMA_F32_16X16X32_F16 (64 K-steps x 4 WMMA)
//   - sum-of-squares for RMSNorm accumulated from the same A loads
//     (4 independent FMA chains to avoid serial VALU_DEP stalls)
//   - per-token rms scale applied to logits post-GEMM (algebraically exact)
//   - top-8 + renormalize (global softmax denominator cancels)

typedef _Float16 v16h __attribute__((ext_vector_type(16)));
typedef _Float16 v8h  __attribute__((ext_vector_type(8)));
typedef _Float16 v4h  __attribute__((ext_vector_type(4)));
typedef float    v8f  __attribute__((ext_vector_type(8)));
typedef float    v4f  __attribute__((ext_vector_type(4)));

#define F_DIM   2048
#define NEXP    64
#define TOPK    8
#define BROW    2056                       // 2048 + 8 halfs pad (bank-conflict-free b128)
#define LSTR    68                         // logits row stride in floats
#define B_BYTES (NEXP * BROW * 2)          // 263168
#define L_BYTES (8 * 16 * LSTR * 4)        // 34816
#define SMEM_BYTES (B_BYTES + L_BYTES)     // 297984 <= 320KB WGP LDS

__global__ __launch_bounds__(256, 1)
void moe_router_fused(const float* __restrict__ x,
                      const float* __restrict__ gw,
                      const float* __restrict__ rs,
                      float* __restrict__ out_w,
                      int*   __restrict__ out_idx)
{
    extern __shared__ char smem[];
    _Float16* Bl = (_Float16*)smem;                 // f16 gate_w * router_scale
    float*    Ll = (float*)(smem + B_BYTES);        // per-wave logit scratch

    const int tid  = (int)threadIdx.x;
    const int lane = tid & 31;
    const int wave = tid >> 5;
    const int n    = lane & 15;      // token row (A) / expert col (B) within tile
    const int hi   = lane >> 4;      // half-wave select

    // ---- stage gate_w * router_scale -> LDS f16, padded rows ----
    for (int i = tid; i < (NEXP * F_DIM) / 4; i += 256) {
        const int f   = i << 2;
        const int row = f >> 11;             // / 2048
        const int col = f & (F_DIM - 1);
        v4f w = *(const v4f*)(gw + f);
        v4f s = *(const v4f*)(rs + col);
        v4h h;
        h[0] = (_Float16)(w[0] * s[0]);
        h[1] = (_Float16)(w[1] * s[1]);
        h[2] = (_Float16)(w[2] * s[2]);
        h[3] = (_Float16)(w[3] * s[3]);
        *(v4h*)(Bl + row * BROW + col) = h;  // 8B aligned
    }

    const int tokBase = (int)blockIdx.x * 128 + wave * 16;
    const float* rowp = x + (size_t)(tokBase + n) * F_DIM;

    __builtin_prefetch(rowp + hi * 1024, 0, 3);   // warm the stream

    __syncthreads();   // B tile visible to all waves

    v8f acc0 = {}, acc1 = {}, acc2 = {}, acc3 = {};
    // 4 independent sum-of-squares chains (merged after the loop)
    float ss0 = 0.0f, ss1 = 0.0f, ss2 = 0.0f, ss3 = 0.0f;

    const _Float16* bcol = Bl + hi * 8;

    #pragma unroll 2
    for (int kc = 0; kc < F_DIM; kc += 32) {
        // unconditional: lanes n and n+16 request the same line, cache dedupes.
        // speculative prefetch past the row end is silently dropped.
        __builtin_prefetch(rowp + kc + 256, 0, 3);   // ~8 iters ahead

        // A fragment: lane holds row n, K = {kc+8*hi+0..7, kc+16+8*hi+0..7}
        v4f a0 = *(const v4f*)(rowp + kc + hi * 8);
        v4f a1 = *(const v4f*)(rowp + kc + hi * 8 + 4);
        v4f a2 = *(const v4f*)(rowp + kc + hi * 8 + 16);
        v4f a3 = *(const v4f*)(rowp + kc + hi * 8 + 20);

        v16h A;
        #pragma unroll
        for (int e = 0; e < 4; ++e) {
            A[e]      = (_Float16)a0[e];
            A[4 + e]  = (_Float16)a1[e];
            A[8 + e]  = (_Float16)a2[e];
            A[12 + e] = (_Float16)a3[e];
            ss0 = fmaf(a0[e], a0[e], ss0);
            ss1 = fmaf(a1[e], a1[e], ss1);
            ss2 = fmaf(a2[e], a2[e], ss2);
            ss3 = fmaf(a3[e], a3[e], ss3);
        }

        // B fragments: lane holds expert (16b + n), K = {kc+8*hi+0..7, kc+16+8*hi+0..7}
        const _Float16* bp = bcol + kc;
        union BU { v16h v; v8h h[2]; } b0, b1, b2, b3;
        b0.h[0] = *(const v8h*)(bp + (n     ) * BROW);
        b0.h[1] = *(const v8h*)(bp + (n     ) * BROW + 16);
        b1.h[0] = *(const v8h*)(bp + (n + 16) * BROW);
        b1.h[1] = *(const v8h*)(bp + (n + 16) * BROW + 16);
        b2.h[0] = *(const v8h*)(bp + (n + 32) * BROW);
        b2.h[1] = *(const v8h*)(bp + (n + 32) * BROW + 16);
        b3.h[0] = *(const v8h*)(bp + (n + 48) * BROW);
        b3.h[1] = *(const v8h*)(bp + (n + 48) * BROW + 16);

        acc0 = __builtin_amdgcn_wmma_f32_16x16x32_f16(false, A, false, b0.v, (short)0, acc0, false, false);
        acc1 = __builtin_amdgcn_wmma_f32_16x16x32_f16(false, A, false, b1.v, (short)0, acc1, false, false);
        acc2 = __builtin_amdgcn_wmma_f32_16x16x32_f16(false, A, false, b2.v, (short)0, acc2, false, false);
        acc3 = __builtin_amdgcn_wmma_f32_16x16x32_f16(false, A, false, b3.v, (short)0, acc3, false, false);
    }

    float ssq = (ss0 + ss1) + (ss2 + ss3);
    // lanes l and l^16 together covered token n's full row
    ssq += __shfl_xor(ssq, 16, 32);
    const float myscale =
        rsqrtf(ssq * (1.0f / F_DIM) + 1e-6f) * 0.022097086912079608f; // * F^-0.5

    // spill unscaled logits: D element v in lane = [row v+8*hi][expert 16b+n]
    float* lw = Ll + wave * 16 * LSTR;
    #pragma unroll
    for (int v = 0; v < 8; ++v) {
        const int m = v + 8 * hi;
        lw[m * LSTR +      n] = acc0[v];
        lw[m * LSTR + 16 + n] = acc1[v];
        lw[m * LSTR + 32 + n] = acc2[v];
        lw[m * LSTR + 48 + n] = acc3[v];
    }

    __syncthreads();

    // ---- top-8 + renormalize; lane t (<16) owns token t; myscale matches (n==lane) ----
    if (lane < 16) {
        const float* lp = Ll + wave * 16 * LSTR + lane * LSTR;
        float lv[NEXP];
        #pragma unroll
        for (int e4 = 0; e4 < 16; ++e4) {
            v4f t = *(const v4f*)(lp + e4 * 4);
            lv[e4 * 4 + 0] = t[0] * myscale;
            lv[e4 * 4 + 1] = t[1] * myscale;
            lv[e4 * 4 + 2] = t[2] * myscale;
            lv[e4 * 4 + 3] = t[3] * myscale;
        }
        unsigned long long used = 0ull;
        float bw[TOPK]; int bi[TOPK];
        #pragma unroll
        for (int j = 0; j < TOPK; ++j) {
            float best = -3.0e38f; int bidx = 0;
            #pragma unroll
            for (int e = 0; e < NEXP; ++e) {
                const bool avail = ((used >> e) & 1ull) == 0ull;
                const float c = avail ? lv[e] : -3.0e38f;
                if (c > best) { best = c; bidx = e; }   // strict > == lower-index tie-break
            }
            used |= (1ull << bidx);
            bw[j] = best; bi[j] = bidx;
        }
        const float mx = bw[0];
        float sum = 0.0f;
        #pragma unroll
        for (int j = 0; j < TOPK; ++j) { bw[j] = __expf(bw[j] - mx); sum += bw[j]; }
        const float inv = 1.0f / fmaxf(sum, 1e-12f);    // softmax Z cancels in the ratio
        const int gtok = tokBase + lane;
        #pragma unroll
        for (int j = 0; j < TOPK; ++j) {
            out_w[gtok * TOPK + j]   = bw[j] * inv;
            out_idx[gtok * TOPK + j] = bi[j];
        }
    }
}

extern "C" void kernel_launch(void* const* d_in, const int* in_sizes, int n_in,
                              void* d_out, int out_size, void* d_ws, size_t ws_size,
                              hipStream_t stream) {
    (void)n_in; (void)out_size; (void)d_ws; (void)ws_size;
    const float* x  = (const float*)d_in[0];   // [B,L,F] fp32
    const float* gw = (const float*)d_in[1];   // [E,F]   fp32
    const float* rs = (const float*)d_in[2];   // [F]     fp32

    const int ntok = in_sizes[0] / F_DIM;      // 32768
    float* out_w   = (float*)d_out;                        // [ntok,8] fp32
    int*   out_idx = (int*)d_out + (size_t)ntok * TOPK;    // [ntok,8] int32

    // >64KB dynamic LDS: raise the cap (host-side attr set; safe outside capture)
    (void)hipFuncSetAttribute((const void*)moe_router_fused,
                              hipFuncAttributeMaxDynamicSharedMemorySize, SMEM_BYTES);

    const int nwg = ntok / 128;                // 128 tokens per 256-thread WG
    moe_router_fused<<<dim3(nwg), dim3(256), SMEM_BYTES, stream>>>(
        x, gw, rs, out_w, out_idx);
}
